// GroupedQueryAttention_44659069944458
// MI455X (gfx1250) — compile-verified, matrix operand order checked by reading, not run on hardware
//
#include <hip/hip_runtime.h>
#include <hip/hip_bf16.h>

// ---------------------------------------------------------------------------
// GQA forward for B=1, S=2048, DIM=2048, NH=32, NKV=8, HD=64 on gfx1250.
// All matmuls run as v_wmma_f32_16x16x32_bf16 (wave32, f32 accumulate).
// V is pre-transposed so both attention matmuls load b128 fragments.
// Causal mask handled in a single specialized diagonal block per query tile.
// Softmax row-sums are computed on the matrix engine (P x ones WMMA).
// ---------------------------------------------------------------------------

typedef __bf16 bf16_t;
typedef __attribute__((ext_vector_type(16))) __bf16 v16bf;
typedef __attribute__((ext_vector_type(8)))  __bf16 v8bf;
typedef __attribute__((ext_vector_type(8)))  float  v8f;

union V16 { v16bf v; v8bf h[2]; };

#define WMMA_BF16(a, b, c) \
  __builtin_amdgcn_wmma_f32_16x16x32_bf16(false, (a), false, (b), (short)0, (c), false, false)

// Load a 16x32 bf16 A/B fragment row for this lane.
// Lane (lane&15) holds the matrix row/col; the lane's 16 values are
// K = hi*8..hi*8+7 and K = 16+hi*8..16+hi*8+7 (hi = lane>>4).
// rowPtr must be 16B aligned.
static __device__ __forceinline__ v16bf load_frag(const bf16_t* rowPtr, int hi) {
  V16 u;
  const v8bf* p = (const v8bf*)rowPtr;
  u.h[0] = p[hi];
  u.h[1] = p[2 + hi];
  return u.v;
}

// ------------------------------- convert -----------------------------------
__global__ __launch_bounds__(256) void f32_to_bf16_kernel(const float* __restrict__ in,
                                                          bf16_t* __restrict__ out, int n) {
  int i = blockIdx.x * 256 + threadIdx.x;
  if (i < n) out[i] = (bf16_t)in[i];
}

// ---------------------------- V transpose ----------------------------------
// vb: [S=2048, 512] -> vt: [512, 2048]  (so PV B-fragments are contiguous)
__global__ __launch_bounds__(256) void transpose_v_kernel(const bf16_t* __restrict__ vb,
                                                          bf16_t* __restrict__ vt) {
  const int idx = blockIdx.x * 256 + threadIdx.x;  // over 2048*512
  const int t = idx >> 9;
  const int c = idx & 511;
  vt[(size_t)c * 2048 + t] = vb[idx];
}

// ------------------------------- GEMM --------------------------------------
// C[M,N] = A[M,K] * W[N,K]^T   (A,W bf16 row-major; C bf16 or f32)
// Block = 256 threads = 8 waves, tile 256x128; wave tile 64x64 (4x4 WMMAs).
template <int OUT_F32>
__global__ __launch_bounds__(256)
void gemm_bf16_wmma(const bf16_t* __restrict__ A, const bf16_t* __restrict__ W,
                    void* __restrict__ Cout, int M, int N, int K) {
  const int lane = threadIdx.x & 31;
  const int wave = threadIdx.x >> 5;
  const int row  = lane & 15;
  const int hi   = lane >> 4;
  const int mBase = blockIdx.y * 256 + (wave >> 1) * 64;
  const int nBase = blockIdx.x * 128 + (wave & 1) * 64;

  const bf16_t* aRow[4];
#pragma unroll
  for (int r = 0; r < 4; ++r)
    aRow[r] = A + (size_t)(mBase + r * 16 + row) * K;
  const bf16_t* wRow[4];
#pragma unroll
  for (int c = 0; c < 4; ++c)
    wRow[c] = W + (size_t)(nBase + c * 16 + row) * K;

  v8f acc[4][4] = {};

  for (int kk = 0; kk < K; kk += 32) {
    v16bf a[4];
#pragma unroll
    for (int r = 0; r < 4; ++r) {
      a[r] = load_frag(aRow[r] + kk, hi);
      if (kk + 64 < K)  // stream next K-chunk (global_prefetch_b8)
        __builtin_prefetch(aRow[r] + kk + 64, 0, 2);
    }
#pragma unroll
    for (int c = 0; c < 4; ++c) {
      v16bf b = load_frag(wRow[c] + kk, hi);
#pragma unroll
      for (int r = 0; r < 4; ++r)
        acc[r][c] = WMMA_BF16(a[r], b, acc[r][c]);
    }
  }

  // C/D layout: VGPR i -> row (i + hi*8), col = lane&15 within each 16x16 tile
#pragma unroll
  for (int r = 0; r < 4; ++r)
#pragma unroll
    for (int c = 0; c < 4; ++c)
#pragma unroll
      for (int i = 0; i < 8; ++i) {
        const int oR = mBase + r * 16 + i + hi * 8;
        const int oC = nBase + c * 16 + row;
        const float val = acc[r][c][i];
        if (OUT_F32) ((float*)Cout)[(size_t)oR * N + oC] = val;
        else         ((bf16_t*)Cout)[(size_t)oR * N + oC] = (bf16_t)val;
      }
}

// ------------------------------- RoPE --------------------------------------
// In-place rotary embedding on a [S, H, 64] bf16 buffer; fc = [S, 32, 2] f32.
__global__ __launch_bounds__(256) void rope_kernel(bf16_t* __restrict__ buf,
                                                   const float* __restrict__ fc, int H) {
  const int idx = blockIdx.x * 256 + threadIdx.x;
  const int d2 = idx & 31;
  const int h  = (idx >> 5) % H;
  const int s  = idx / (H * 32);
  const float c  = fc[s * 64 + d2 * 2 + 0];
  const float sn = fc[s * 64 + d2 * 2 + 1];
  const size_t base = (size_t)s * (H * 64) + h * 64 + 2 * d2;
  const float x1 = (float)buf[base], x2 = (float)buf[base + 1];
  buf[base]     = (bf16_t)(x1 * c - x2 * sn);
  buf[base + 1] = (bf16_t)(x1 * sn + x2 * c);
}

// ------------------------- flash GQA attention -----------------------------
// One wave handles one (head, 16-row query tile). Online softmax over key
// blocks of 32. Per block: 4 QK WMMAs, 4 PV WMMAs, and 1 "P x ones" WMMA
// that produces the softmax row-sums lane-replicated in accumulator layout
// (replacing a 4-step ds_bpermute butterfly). The causal mask is applied
// only in the single diagonal block; Q is pre-scaled by 1/sqrt(HD).
template <bool MASKED>
static __device__ __forceinline__ void attn_block(
    int t0, int qt, int row, int hi, int kvh,
    const bf16_t* __restrict__ Kb, const bf16_t* const vCol[4],
    bf16_t* __restrict__ plsW, v16bf aQ0, v16bf aQ1, v16bf ones,
    v8f oacc[4], v8f& lacc, float mrow[8]) {
  constexpr int NKVD = 512, HD = 64;

  // ---- S = Qs * K^T for 2 x 16 keys ----
  v8f st[2] = {};
#pragma unroll
  for (int nb = 0; nb < 2; ++nb) {
    const int key = t0 + nb * 16 + row;  // always < S
    const bf16_t* kRow = Kb + (size_t)key * NKVD + kvh * HD;
    v16bf bK0 = load_frag(kRow + 0, hi);
    v16bf bK1 = load_frag(kRow + 32, hi);
    st[nb] = WMMA_BF16(aQ0, bK0, st[nb]);
    st[nb] = WMMA_BF16(aQ1, bK1, st[nb]);
  }

  // ---- online softmax: row-max butterfly + exp; sums via WMMA below ----
#pragma unroll
  for (int i = 0; i < 8; ++i) {
    float m_i = -1e30f;
    float p0 = st[0][i], p1 = st[1][i];
    if (MASKED) {
      const int r = qt * 16 + i + hi * 8;
      if (t0 + row > r)      p0 = -1e30f;  // causal
      if (t0 + 16 + row > r) p1 = -1e30f;
    }
    m_i = fmaxf(p0, p1);
#pragma unroll
    for (int off = 1; off < 16; off <<= 1)
      m_i = fmaxf(m_i, __shfl_xor(m_i, off, 16));
    const float m_new = fmaxf(mrow[i], m_i);
    const float alpha = __expf(mrow[i] - m_new);
    mrow[i] = m_new;
    lacc[i] *= alpha;
#pragma unroll
    for (int nt = 0; nt < 4; ++nt) oacc[nt][i] *= alpha;
    // bounce P through LDS to re-layout C-tile -> A-fragment
    plsW[(i + hi * 8) * 32 + row]      = (bf16_t)__expf(p0 - m_new);
    plsW[(i + hi * 8) * 32 + 16 + row] = (bf16_t)__expf(p1 - m_new);
  }
  asm volatile("s_wait_dscnt 0" ::: "memory");

  V16 aP;
  {
    const v8bf* pp = (const v8bf*)&plsW[row * 32];
    aP.h[0] = pp[hi];
    aP.h[1] = pp[2 + hi];
  }

  // ---- l += P * 1 : row-sums, lane-replicated, same alpha recurrence ----
  lacc = WMMA_BF16(aP.v, ones, lacc);

  // ---- O += P(16x32) * V(32x64), B-fragments contiguous from V^T ----
#pragma unroll
  for (int nt = 0; nt < 4; ++nt) {
    v16bf bv = load_frag(vCol[nt] + t0, hi);
    oacc[nt] = WMMA_BF16(aP.v, bv, oacc[nt]);
  }
}

__global__ __launch_bounds__(128)
void gqa_attention(const bf16_t* __restrict__ Q, const bf16_t* __restrict__ Kb,
                   const bf16_t* __restrict__ Vt, bf16_t* __restrict__ O) {
  constexpr int S = 2048, NQD = 2048, HD = 64;
  __shared__ bf16_t pls[4][16 * 32];  // per-wave P-tile bounce buffer

  const int lane = threadIdx.x & 31;
  const int wave = threadIdx.x >> 5;
  const int gw   = blockIdx.x * 4 + wave;
  const int head = gw >> 7;   // / (S/16)
  const int qt   = gw & 127;
  const int kvh  = head >> 2; // NREP = 4
  const int row  = lane & 15;
  const int hi   = lane >> 4;

  const bf16_t* qRow = Q + (size_t)(qt * 16 + row) * NQD + head * HD;
  v16bf aQ0 = load_frag(qRow + 0, hi);   // HD 0..31
  v16bf aQ1 = load_frag(qRow + 32, hi);  // HD 32..63
  v16bf ones;
  // Fold softmax scale 1/sqrt(64) into Q once (exact exponent shift in bf16),
  // and build the all-ones B fragment for the row-sum WMMA.
#pragma unroll
  for (int j = 0; j < 16; ++j) {
    aQ0[j] = (bf16_t)((float)aQ0[j] * 0.125f);
    aQ1[j] = (bf16_t)((float)aQ1[j] * 0.125f);
    ones[j] = (bf16_t)1.0f;
  }

  // Per-lane V^T column pointers for the 4 output N-tiles.
  const bf16_t* vCol[4];
#pragma unroll
  for (int nt = 0; nt < 4; ++nt)
    vCol[nt] = Vt + (size_t)(kvh * HD + nt * 16 + row) * S;

  v8f oacc[4] = {};
  v8f lacc = {};
  float mrow[8];
#pragma unroll
  for (int i = 0; i < 8; ++i) mrow[i] = -1e30f;

  bf16_t* plsW = pls[wave];

  // Unmasked main loop: all blocks strictly below the diagonal.
  const int tDiag = 32 * (qt >> 1);
  for (int t0 = 0; t0 < tDiag; t0 += 32)
    attn_block<false>(t0, qt, row, hi, kvh, Kb, vCol, plsW, aQ0, aQ1, ones,
                      oacc, lacc, mrow);
  // Single diagonal block with causal masking.
  attn_block<true>(tDiag, qt, row, hi, kvh, Kb, vCol, plsW, aQ0, aQ1, ones,
                   oacc, lacc, mrow);

  // ---- normalize + store attn output (bf16) ----
#pragma unroll
  for (int nt = 0; nt < 4; ++nt)
#pragma unroll
    for (int i = 0; i < 8; ++i) {
      const int r = qt * 16 + i + hi * 8;
      O[(size_t)r * NQD + head * HD + nt * 16 + row] =
          (bf16_t)(oacc[nt][i] / lacc[i]);
    }
}

// ------------------------------- launcher ----------------------------------
extern "C" void kernel_launch(void* const* d_in, const int* in_sizes, int n_in,
                              void* d_out, int out_size, void* d_ws, size_t ws_size,
                              hipStream_t stream) {
  constexpr int S = 2048, DIM = 2048, NH = 32, NKV = 8, HD = 64;
  constexpr int NQ = NH * HD;     // 2048
  constexpr int NKVD = NKV * HD;  // 512

  const float* x  = (const float*)d_in[0];
  const float* wq = (const float*)d_in[1];
  const float* wk = (const float*)d_in[2];
  const float* wv = (const float*)d_in[3];
  const float* wo = (const float*)d_in[4];
  const float* fc = (const float*)d_in[5];
  // d_in[6] = mask: exactly causal, applied analytically in-kernel.

  char* ws = (char*)d_ws;
  size_t off = 0;
  bf16_t* xb  = (bf16_t*)(ws + off); off += (size_t)S * DIM * 2;
  bf16_t* wqb = (bf16_t*)(ws + off); off += (size_t)NQ * DIM * 2;
  bf16_t* wkb = (bf16_t*)(ws + off); off += (size_t)NKVD * DIM * 2;
  bf16_t* wvb = (bf16_t*)(ws + off); off += (size_t)NKVD * DIM * 2;
  bf16_t* wob = (bf16_t*)(ws + off); off += (size_t)DIM * NQ * 2;
  bf16_t* qb  = (bf16_t*)(ws + off); off += (size_t)S * NQ * 2;
  bf16_t* kb  = (bf16_t*)(ws + off); off += (size_t)S * NKVD * 2;
  bf16_t* vb  = (bf16_t*)(ws + off); off += (size_t)S * NKVD * 2;
  bf16_t* vt  = (bf16_t*)(ws + off); off += (size_t)NKVD * S * 2;
  bf16_t* ab  = (bf16_t*)(ws + off); off += (size_t)S * NQ * 2;

  // 1) convert activations + weights to bf16 (compute-bound op -> 2x math rate)
  f32_to_bf16_kernel<<<(S * DIM) / 256, 256, 0, stream>>>(x, xb, S * DIM);
  f32_to_bf16_kernel<<<(NQ * DIM) / 256, 256, 0, stream>>>(wq, wqb, NQ * DIM);
  f32_to_bf16_kernel<<<(NKVD * DIM) / 256, 256, 0, stream>>>(wk, wkb, NKVD * DIM);
  f32_to_bf16_kernel<<<(NKVD * DIM) / 256, 256, 0, stream>>>(wv, wvb, NKVD * DIM);
  f32_to_bf16_kernel<<<(DIM * NQ) / 256, 256, 0, stream>>>(wo, wob, DIM * NQ);

  // 2) QKV projections
  gemm_bf16_wmma<0><<<dim3(NQ / 128, S / 256), 256, 0, stream>>>(xb, wqb, qb, S, NQ, DIM);
  gemm_bf16_wmma<0><<<dim3(NKVD / 128, S / 256), 256, 0, stream>>>(xb, wkb, kb, S, NKVD, DIM);
  gemm_bf16_wmma<0><<<dim3(NKVD / 128, S / 256), 256, 0, stream>>>(xb, wvb, vb, S, NKVD, DIM);

  // 3) RoPE on q and k; transpose V for contiguous PV fragments
  rope_kernel<<<(S * NH * (HD / 2)) / 256, 256, 0, stream>>>(qb, fc, NH);
  rope_kernel<<<(S * NKV * (HD / 2)) / 256, 256, 0, stream>>>(kb, fc, NKV);
  transpose_v_kernel<<<(S * NKVD) / 256, 256, 0, stream>>>(vb, vt);

  // 4) flash GQA attention: NH * (S/16) waves, 4 waves/block
  gqa_attention<<<(NH * (S / 16)) / 4, 128, 0, stream>>>(qb, kb, vt, ab);

  // 5) output projection -> f32
  gemm_bf16_wmma<1><<<dim3(DIM / 128, S / 256), 256, 0, stream>>>(ab, wob, d_out, S, DIM, NQ);

  (void)in_sizes; (void)n_in; (void)out_size; (void)ws_size;
}
